// NPAModel_25391846654581
// MI455X (gfx1250) — matrix operand order repaired and checked by reading, not compiled, not run on hardware
//
#include <hip/hip_runtime.h>

// NPA news-recommendation forward pass for gfx1250 (MI455X), wave32.
// B=64, NPR=4, HIS=50, T=30, V=50000, E=300, F=400, UD=50, PD=200.

#define B_   64
#define NPR_ 4
#define HIS_ 50
#define T_   30
#define V_   50000
#define E_   300
#define F_   400
#define UD_  50
#define PD_  200

#define K3E   (3 * E_)   // 900
#define KPAD  928        // 900 padded to 29 * 32
#define NCOL  32         // T=30 padded to 2 N-tiles of 16
#define FPAD  512        // 400 padded to 32 M-tiles -> 4 per wave, branch-free
#define MT_W  4          // M-tiles per wave

typedef _Float16 v8h  __attribute__((ext_vector_type(8)));
typedef _Float16 v16h __attribute__((ext_vector_type(16)));
typedef float    v8f  __attribute__((ext_vector_type(8)));

// ---------------------------------------------------------------------------
// Kernel 0: convert conv_w [F,E,3] fp32 -> f16, padded row-major [FPAD, KPAD].
// Rows 400..511 and k 900..927 are zeros (consumed by pad WMMAs, discarded).
// ---------------------------------------------------------------------------
__global__ __launch_bounds__(256) void npa_convert_w(
    const float* __restrict__ conv_w, _Float16* __restrict__ cw16)
{
  int idx = blockIdx.x * 256 + threadIdx.x;
  if (idx >= FPAD * KPAD) return;
  int f = idx / KPAD, k = idx - f * KPAD;
  float v = (f < F_ && k < K3E) ? conv_w[f * K3E + k] : 0.0f;
  cw16[idx] = (_Float16)v;
}

// ---------------------------------------------------------------------------
// Kernel 1: user projections -> word_pref[B,F], news_pref[B,F]. Tiny.
// ---------------------------------------------------------------------------
__global__ __launch_bounds__(256) void npa_prefs(
    const float* __restrict__ user_index,
    const float* __restrict__ W_user, const float* __restrict__ b_user,
    const float* __restrict__ W_wq,   const float* __restrict__ b_wq,
    const float* __restrict__ W_nq,   const float* __restrict__ b_nq,
    const float* __restrict__ W_wp,   const float* __restrict__ b_wp,
    const float* __restrict__ W_np,   const float* __restrict__ b_np,
    float* __restrict__ word_pref,    float* __restrict__ news_pref)
{
  __shared__ float eu[UD_];
  __shared__ float wq[PD_];
  __shared__ float nq[PD_];
  const int b = blockIdx.x, tid = threadIdx.x;
  const float ui = user_index[b];

  if (tid < UD_) eu[tid] = ui * W_user[tid] + b_user[tid];   // W_user is [UD,1]
  __syncthreads();

  if (tid < PD_) {
    float s1 = b_wq[tid], s2 = b_nq[tid];
    #pragma unroll 5
    for (int u = 0; u < UD_; ++u) {
      float e = eu[u];
      s1 += e * W_wq[tid * UD_ + u];
      s2 += e * W_nq[tid * UD_ + u];
    }
    wq[tid] = s1 > 0.f ? s1 : 0.f;
    nq[tid] = s2 > 0.f ? s2 : 0.f;
  }
  __syncthreads();

  for (int f = tid; f < F_; f += 256) {
    float s1 = b_wp[f], s2 = b_np[f];
    #pragma unroll 4
    for (int p = 0; p < PD_; ++p) {
      s1 += wq[p] * W_wp[f * PD_ + p];
      s2 += nq[p] * W_np[f * PD_ + p];
    }
    word_pref[b * F_ + f] = s1;
    news_pref[b * F_ + f] = s2;
  }
}

// ---------------------------------------------------------------------------
// Kernel 2 (hot): per news item, conv-as-GEMM with WMMA f16->f32, bias+ReLU,
// word-level attention -> repr[item, F].  One block (8 waves) per item.
//   Y[FPAD, NCOL] = cw16[FPAD, KPAD] x patch[KPAD, NCOL]
// Branch-free GEMM: every wave owns exactly MT_W=4 M-tiles x 2 N-tiles.
// Per K step: 2 B fragments from LDS (shared across M-tiles) + 1 A fragment
// per M-tile (shared across both N-tiles) -> 2 WMMAs per A load, 8 total.
// ---------------------------------------------------------------------------
__global__ __launch_bounds__(256) void npa_encode(
    const int*      __restrict__ tokens, int n_per_b,
    const float*    __restrict__ embedding,
    const _Float16* __restrict__ cw16,
    const float*    __restrict__ conv_b,
    const float*    __restrict__ word_pref,
    float*          __restrict__ out_repr)
{
  __shared__ __align__(16) unsigned char smem[NCOL * KPAD * 2];   // 59392 B
  _Float16* patch = (_Float16*)smem;                  // [NCOL][KPAD] f16
  float*    Ybuf  = (float*)smem;                     // [F_][NCOL] f32 (after barrier)
  float*    aux   = (float*)(smem + F_ * NCOL * 4);   // 32 floats of scratch

  const int item = blockIdx.x;
  const int b    = item / n_per_b;
  const int tid  = threadIdx.x;
  const int lane = tid & 31;
  const int wave = tid >> 5;

  // ---- zero the patch (K pad + column pad must be 0) ----
  for (int i = tid; i < (NCOL * KPAD) / 2; i += 256)
    ((unsigned int*)smem)[i] = 0u;
  __syncthreads();

  // ---- im2col gather: patch[t][e*3+k] = emb[tok[t+k-1]][e], zero-padded ----
  const int* tok = tokens + item * T_;
  for (int idx = tid; idx < T_ * E_; idx += 256) {
    int p = idx / E_, e = idx - p * E_;
    _Float16 h = (_Float16)embedding[(long)tok[p] * E_ + e];
    #pragma unroll
    for (int k = 0; k < 3; ++k) {
      int c = p + 1 - k;                 // column that consumes token p at tap k
      if (c >= 0 && c < T_) patch[c * KPAD + e * 3 + k] = h;
    }
  }
  __syncthreads();

  // ---- GEMM: 32 M-tiles x 2 N-tiles, 8 waves, 4 M-tiles each, no branches --
  v8f acc[MT_W][2];
  #pragma unroll
  for (int i = 0; i < MT_W; ++i) {
    acc[i][0] = (v8f){0,0,0,0,0,0,0,0};
    acc[i][1] = (v8f){0,0,0,0,0,0,0,0};
  }

  const int aoff = (lane & 16) >> 1;  // A frag: lanes 16-31 start at K+8
  const int boff = (lane & 16) >> 1;  // B frag: lanes 16-31 start at K+8

  const _Float16* Ab[MT_W];
  #pragma unroll
  for (int i = 0; i < MT_W; ++i) {
    int mt = wave + i * 8;            // 0..31, always valid (FPAD rows)
    Ab[i] = cw16 + (mt * 16 + (lane & 15)) * KPAD + aoff;
  }
  const _Float16* B0 = patch + (lane & 15) * KPAD + boff;         // nt = 0
  const _Float16* B1 = patch + (16 + (lane & 15)) * KPAD + boff;  // nt = 1

  for (int kt = 0; kt < KPAD / 32; ++kt) {
    int k0 = kt * 32;
    // B 32x16 f16: per lane K chunks {k0..+7, k0+16..+23} (+8 upper half)
    v8h b0lo = *(const v8h*)(B0 + k0);
    v8h b0hi = *(const v8h*)(B0 + k0 + 16);
    v8h b1lo = *(const v8h*)(B1 + k0);
    v8h b1hi = *(const v8h*)(B1 + k0 + 16);
    v16h bb0 = __builtin_shufflevector(b0lo, b0hi, 0,1,2,3,4,5,6,7,8,9,10,11,12,13,14,15);
    v16h bb1 = __builtin_shufflevector(b1lo, b1hi, 0,1,2,3,4,5,6,7,8,9,10,11,12,13,14,15);
    #pragma unroll
    for (int i = 0; i < MT_W; ++i) {
      // A 16x32 f16: per lane K chunks {k0..+7, k0+16..+23} (+8 upper half)
      v8h alo = *(const v8h*)(Ab[i] + k0);
      v8h ahi = *(const v8h*)(Ab[i] + k0 + 16);
      v16h a  = __builtin_shufflevector(alo, ahi, 0,1,2,3,4,5,6,7,8,9,10,11,12,13,14,15);
      acc[i][0] = __builtin_amdgcn_wmma_f32_16x16x32_f16(
                      false, a, false, bb0, (short)0, acc[i][0], false, false);
      acc[i][1] = __builtin_amdgcn_wmma_f32_16x16x32_f16(
                      false, a, false, bb1, (short)0, acc[i][1], false, false);
    }
  }
  __syncthreads();   // everyone done reading patch; LDS now becomes Ybuf

  // ---- bias + ReLU, scatter fragments to Ybuf[f][t]; skip pad rows ----
  {
    const int half = lane >> 4;        // D: lanes 16-31 hold M=j+8
    const int n    = lane & 15;
    #pragma unroll
    for (int i = 0; i < MT_W; ++i) {
      int mt = wave + i * 8;
      if (mt >= F_ / 16) continue;     // pad M-tiles: results are zeros, drop
      #pragma unroll
      for (int nt = 0; nt < 2; ++nt) {
        int t = nt * 16 + n;
        #pragma unroll
        for (int j = 0; j < 8; ++j) {
          int f = mt * 16 + j + half * 8;
          float y = acc[i][nt][j] + conv_b[f];
          Ybuf[f * NCOL + t] = y > 0.f ? y : 0.f;
        }
      }
    }
  }
  __syncthreads();

  // ---- word attention logits: sc[t] = <word_pref[b], Y[:,t]> ----
  const float* wp = word_pref + b * F_;
  {
    int t = tid >> 3, l8 = tid & 7;    // 8 lanes per column
    float part = 0.f;
    if (t < T_)
      for (int f = l8; f < F_; f += 8) part += wp[f] * Ybuf[f * NCOL + t];
    part += __shfl_down(part, 4, 8);
    part += __shfl_down(part, 2, 8);
    part += __shfl_down(part, 1, 8);
    if (l8 == 0 && t < T_) aux[t] = part;
  }
  __syncthreads();

  // ---- softmax over T=30 (wave 0) ----
  if (wave == 0) {
    float x = (lane < T_) ? aux[lane] : -3.4e38f;
    float m = x;
    for (int off = 16; off; off >>= 1) m = fmaxf(m, __shfl_xor(m, off, 32));
    float e = (lane < T_) ? __expf(x - m) : 0.f;
    float s = e;
    for (int off = 16; off; off >>= 1) s += __shfl_xor(s, off, 32);
    if (lane < T_) aux[lane] = e / s;
  }
  __syncthreads();

  // ---- repr[f] = sum_t Y[f,t] * attn[t] ----
  float* orow = out_repr + (long)item * F_;
  for (int f = tid; f < F_; f += 256) {
    float r = 0.f;
    #pragma unroll 6
    for (int t = 0; t < T_; ++t) r += Ybuf[f * NCOL + t] * aux[t];
    orow[f] = r;
  }
}

// ---------------------------------------------------------------------------
// Kernel 3: history attention, user repr, candidate scores, log_softmax.
// ---------------------------------------------------------------------------
__global__ __launch_bounds__(256) void npa_score(
    const float* __restrict__ news_pref,
    const float* __restrict__ cdd_repr,
    const float* __restrict__ his_repr,
    float* __restrict__ out)
{
  __shared__ float snp[F_];
  __shared__ float sur[F_];
  __shared__ float saw[64];
  __shared__ float ss[8];
  const int b = blockIdx.x, tid = threadIdx.x, lane = tid & 31, wave = tid >> 5;

  for (int f = tid; f < F_; f += 256) snp[f] = news_pref[b * F_ + f];
  __syncthreads();

  // logits over history: saw[h] = <news_pref[b], his_repr[b,h,:]>
  {
    int g = tid >> 3, l8 = tid & 7;
    for (int h = g; h < HIS_; h += 32) {
      const float* hr = his_repr + ((long)b * HIS_ + h) * F_;
      float p = 0.f;
      for (int f = l8; f < F_; f += 8) p += snp[f] * hr[f];
      p += __shfl_down(p, 4, 8);
      p += __shfl_down(p, 2, 8);
      p += __shfl_down(p, 1, 8);
      if (l8 == 0) saw[h] = p;
    }
  }
  __syncthreads();

  // softmax over HIS=50 (wave 0, 2 elements per lane)
  if (wave == 0) {
    float x0 = (lane < HIS_) ? saw[lane] : -3.4e38f;
    float x1 = (lane + 32 < HIS_) ? saw[lane + 32] : -3.4e38f;
    float m = fmaxf(x0, x1);
    for (int off = 16; off; off >>= 1) m = fmaxf(m, __shfl_xor(m, off, 32));
    float e0 = (lane < HIS_) ? __expf(x0 - m) : 0.f;
    float e1 = (lane + 32 < HIS_) ? __expf(x1 - m) : 0.f;
    float s = e0 + e1;
    for (int off = 16; off; off >>= 1) s += __shfl_xor(s, off, 32);
    if (lane < HIS_) saw[lane] = e0 / s;
    if (lane + 32 < HIS_) saw[lane + 32] = e1 / s;
  }
  __syncthreads();

  // user_repr[f] = sum_h his_repr[b,h,f] * aw[h]
  for (int f = tid; f < F_; f += 256) {
    float r = 0.f;
    #pragma unroll 5
    for (int h = 0; h < HIS_; ++h)
      r += his_repr[((long)b * HIS_ + h) * F_ + f] * saw[h];
    sur[f] = r;
  }
  __syncthreads();

  // scores: one wave per candidate (waves 0..4 fully active)
  if (tid < 32 * (NPR_ + 1)) {
    int n = tid >> 5, l = tid & 31;
    const float* cr = cdd_repr + ((long)b * (NPR_ + 1) + n) * F_;
    float p = 0.f;
    for (int f = l; f < F_; f += 32) p += cr[f] * sur[f];
    for (int off = 16; off; off >>= 1) p += __shfl_xor(p, off, 32);
    if (l == 0) ss[n] = p;
  }
  __syncthreads();

  if (tid == 0) {
    float m = ss[0];
    for (int n = 1; n <= NPR_; ++n) m = fmaxf(m, ss[n]);
    float s = 0.f;
    for (int n = 0; n <= NPR_; ++n) s += __expf(ss[n] - m);
    float ls = __logf(s);
    for (int n = 0; n <= NPR_; ++n) out[b * (NPR_ + 1) + n] = ss[n] - m - ls;
  }
}

// ---------------------------------------------------------------------------
// Workspace layout (bytes):
//   cw16      : 0        .. 950272   (512*928 f16, zero-padded)
//   word_pref : 950272   .. 1052672  (64*400 f32)
//   news_pref : 1052672  .. 1155072
//   cdd_repr  : 1155072  .. 1667072  (64*5*400 f32)
//   his_repr  : 1667072  .. 6787072  (64*50*400 f32)
// ---------------------------------------------------------------------------
extern "C" void kernel_launch(void* const* d_in, const int* in_sizes, int n_in,
                              void* d_out, int out_size, void* d_ws, size_t ws_size,
                              hipStream_t stream)
{
  const float* user_index = (const float*)d_in[0];
  const int*   cand       = (const int*)  d_in[1];
  const int*   clicked    = (const int*)  d_in[2];
  const float* embedding  = (const float*)d_in[3];
  const float* W_user     = (const float*)d_in[4];
  const float* b_user     = (const float*)d_in[5];
  const float* W_wq       = (const float*)d_in[6];
  const float* b_wq       = (const float*)d_in[7];
  const float* W_nq       = (const float*)d_in[8];
  const float* b_nq       = (const float*)d_in[9];
  const float* W_wp       = (const float*)d_in[10];
  const float* b_wp       = (const float*)d_in[11];
  const float* W_np       = (const float*)d_in[12];
  const float* b_np       = (const float*)d_in[13];
  const float* conv_w     = (const float*)d_in[14];
  const float* conv_b     = (const float*)d_in[15];

  char* ws = (char*)d_ws;
  _Float16* cw16      = (_Float16*)(ws + 0);
  float*    word_pref = (float*)(ws + 950272);
  float*    news_pref = (float*)(ws + 1052672);
  float*    cdd_repr  = (float*)(ws + 1155072);
  float*    his_repr  = (float*)(ws + 1667072);

  npa_convert_w<<<(FPAD * KPAD + 255) / 256, 256, 0, stream>>>(conv_w, cw16);

  npa_prefs<<<B_, 256, 0, stream>>>(user_index, W_user, b_user,
                                    W_wq, b_wq, W_nq, b_nq,
                                    W_wp, b_wp, W_np, b_np,
                                    word_pref, news_pref);

  npa_encode<<<B_ * (NPR_ + 1), 256, 0, stream>>>(
      cand, NPR_ + 1, embedding, cw16, conv_b, word_pref, cdd_repr);

  npa_encode<<<B_ * HIS_, 256, 0, stream>>>(
      clicked, HIS_, embedding, cw16, conv_b, word_pref, his_repr);

  npa_score<<<B_, 256, 0, stream>>>(news_pref, cdd_repr, his_repr, (float*)d_out);
}
